// StackRNN_36163624632543
// MI455X (gfx1250) — matrix operand order repaired and checked by reading, not compiled, not run on hardware
//
#include <hip/hip_runtime.h>

// ---- problem constants (match reference) ----
#define BTOT   256
#define TLEN   512
#define NCHAR  128
#define NHID   256
#define NSTACK 10
#define SSIZE  200

#define BW   16                 // batch rows per workgroup (= WMMA M)
#define NWG  (BTOT / BW)        // 16 workgroups
#define SHS  264                // s_h row stride (f16) -> conflict-friendly

typedef __attribute__((ext_vector_type(16))) _Float16 v16h;
typedef __attribute__((ext_vector_type(8)))  _Float16 v8h;
typedef __attribute__((ext_vector_type(4)))  _Float16 v4h;
typedef __attribute__((ext_vector_type(8)))  float    v8f;
typedef __attribute__((ext_vector_type(4)))  unsigned int v4u;
typedef __attribute__((ext_vector_type(8)))  int      v8i;
typedef __attribute__((ext_vector_type(4)))  int      v4i;

#if __has_builtin(__builtin_amdgcn_tensor_load_to_lds)
#define TDM_OK 1
#endif

// ---- A fragment (16x32 f16) from row-major LDS [16][stride] ----
// lane l: M = l&15 ; K = K0..K0+15 contiguous, K0 = (l&16)?16:0
__device__ __forceinline__ v16h loadA(const _Float16* base, int lane, int kbase, int stride) {
  const int m  = lane & 15;
  const int k0 = (lane & 16) ? 16 : 0;
  const _Float16* p = base + m * stride + kbase + k0;
  v8h lo = *(const v8h*)(p);
  v8h hi = *(const v8h*)(p + 8);
  v16h a;
#pragma unroll
  for (int q = 0; q < 8; ++q) { a[q] = lo[q]; a[8 + q] = hi[q]; }
  return a;
}

// ---- B fragment: pre-packed, 16 f16 contiguous per lane ----
__device__ __forceinline__ v16h loadBfrag(const _Float16* fragbase, int lane) {
  const _Float16* p = fragbase + lane * 16;
  v8h lo = *(const v8h*)(p);
  v8h hi = *(const v8h*)(p + 8);
  v16h b;
#pragma unroll
  for (int q = 0; q < 8; ++q) { b[q] = lo[q]; b[8 + q] = hi[q]; }
  return b;
}

// fast sigmoid: v_exp_f32 + v_rcp_f32 (avoid IEEE div expansion)
__device__ __forceinline__ float sigm(float z) {
  return __builtin_amdgcn_rcpf(1.f + __expf(-z));
}

// ---- pre-kernel 1: pack hid2out into WMMA B-fragment layout in d_ws ----
// frag flat index: ((kt*8 + nt)*32 + lane)*16 + i ; k = kt*32 + ((lane&16)?16:0) + i ; n = nt*16 + (lane&15)
__global__ void stackrnn_pack_wout(const float* __restrict__ hid2out, _Float16* __restrict__ wsf) {
  const int e    = blockIdx.x * 256 + threadIdx.x;   // 32768 elems
  const int i    = e & 15;
  const int lane = (e >> 4) & 31;
  const int nt   = (e >> 9) & 7;
  const int kt   = (e >> 12) & 7;
  const int k = kt * 32 + ((lane & 16) ? 16 : 0) + i;
  const int n = nt * 16 + (lane & 15);
  wsf[e] = (_Float16)hid2out[n * NHID + k];          // W_out[k][n] = hid2out[n][k]
}

// ---- pre-kernel 2: transpose in2hid -> embT [NCHAR][NHID] so TDM gather rows are contiguous ----
__global__ void stackrnn_pack_embT(const float* __restrict__ in2hid, float* __restrict__ embT) {
  const int e = blockIdx.x * 256 + threadIdx.x;      // 32768 elems
  const int c = e >> 8, h = e & 255;
  embT[c * NHID + h] = in2hid[h * NCHAR + c];
}

// ---- main scan kernel: one WG = 16 batch rows, full T recurrence ----
__global__ __launch_bounds__(256) void stackrnn_scan(
    const int*   __restrict__ x,          // [B,T]
    const float* __restrict__ in2hid,     // [NHID,NCHAR]
    const float* __restrict__ hid2act,    // [NSTACK,2,NHID]
    const float* __restrict__ hid2stack,  // [NSTACK,SSIZE,NHID]
    const float* __restrict__ stack2hid,  // [NSTACK,NHID,SSIZE]
    const _Float16* __restrict__ woutf,   // packed frags (d_ws)
    const float* __restrict__ embT,       // transposed in2hid (d_ws)
    float* __restrict__ out)              // [B,T,NCHAR]
{
  __shared__ __attribute__((aligned(8)))  _Float16 s_stacks[BW * NSTACK * SSIZE]; // 62.5 KB
  __shared__ __attribute__((aligned(32))) _Float16 s_Wsc [16 * 32 * 16];  // 16 KB  B frags, K=32
  __shared__ __attribute__((aligned(32))) _Float16 s_Wact[8 * 2 * 32 * 16];// 16 KB  B frags, K=256,N=32
  __shared__ __attribute__((aligned(32))) _Float16 s_Wv  [8 * 32 * 16];   // 8 KB   B frags, K=256,N=16
  __shared__ __attribute__((aligned(16))) _Float16 s_h   [BW * SHS];      // h (f16), A operand
  __shared__ __attribute__((aligned(16))) _Float16 s_A   [BW * 32];       // stack-top A operand
  __shared__ float s_act[BW * 32];                                        // softmaxed actions
  __shared__ float s_v  [BW * 16];                                        // push values
  __shared__ __attribute__((aligned(16))) float s_emb[2][BW * NHID];      // emb double buffer

  const int tid  = threadIdx.x;
  const int lane = tid & 31;
  const int wv   = tid >> 5;            // wave id 0..7
  const int b0   = blockIdx.x * BW;

  // ---------------- one-time setup ----------------
  for (int e = tid; e < BW * NSTACK * SSIZE; e += 256) s_stacks[e] = (_Float16)0.f;
  if (tid < BW * 32 / 2) { s_A[tid * 2] = (_Float16)0.f; s_A[tid * 2 + 1] = (_Float16)0.f; }

  // stack2hid -> B frags: K=ks (20 of 32 valid), N=h
  for (int e = tid; e < 16 * 32 * 16; e += 256) {
    const int i = e & 15, ln = (e >> 4) & 31, nt = e >> 9;
    const int ks = ((ln & 16) ? 16 : 0) + i;
    const int h  = nt * 16 + (ln & 15);
    float v = 0.f;
    if (ks < 2 * NSTACK) v = stack2hid[(ks >> 1) * (NHID * SSIZE) + h * SSIZE + (ks & 1)];
    s_Wsc[e] = (_Float16)v;
  }
  // hid2act -> B frags: K=h, N=k*2+a (20 of 32 valid)
  for (int e = tid; e < 8 * 2 * 32 * 16; e += 256) {
    const int i = e & 15, ln = (e >> 4) & 31, nt = (e >> 9) & 1, kt = e >> 10;
    const int h = kt * 32 + ((ln & 16) ? 16 : 0) + i;
    const int n = nt * 16 + (ln & 15);
    float v = 0.f;
    if (n < 2 * NSTACK) v = hid2act[n * NHID + h];
    s_Wact[e] = (_Float16)v;
  }
  // hid2stack top row -> B frags: K=h, N=k (10 of 16 valid)
  for (int e = tid; e < 8 * 32 * 16; e += 256) {
    const int i = e & 15, ln = (e >> 4) & 31, kt = e >> 9;
    const int h  = kt * 32 + ((ln & 16) ? 16 : 0) + i;
    const int kk = ln & 15;
    float v = 0.f;
    if (kk < NSTACK) v = hid2stack[kk * (SSIZE * NHID) + h];
    s_Wv[e] = (_Float16)v;
  }
  // emb gather for t = 0 (manual; TDM takes over inside the loop)
  for (int e = tid; e < BW * NHID; e += 256) {
    const int m = e >> 8, h = e & 255;
    const int c = x[(b0 + m) * TLEN + 0];
    s_emb[0][m * NHID + h] = in2hid[h * NCHAR + c];
  }
  __syncthreads();

  // ---------------- scan (3 barriers per step) ----------------
  for (int t = 0; t < TLEN; ++t) {
    const int eb = t & 1;

    // (B) h = sigmoid(emb + A x Wsc) ; each wave does 2 of 16 N-tiles.
    //     s_A was produced by the previous step's stack update (zeros at t=0).
    {
      const v16h a = loadA(s_A, lane, 0, 32);
      const int n  = lane & 15;
      const int mh = (lane >> 4) << 3;
#pragma unroll
      for (int j = 0; j < 2; ++j) {
        const int nt = wv * 2 + j;
        const v16h b = loadBfrag(&s_Wsc[nt * 32 * 16], lane);
        v8f acc = {};
        acc = __builtin_amdgcn_wmma_f32_16x16x32_f16(false, a, false, b, (short)0, acc, false, false);
#pragma unroll
        for (int r = 0; r < 8; ++r) {
          const int m   = r + mh;
          const int col = nt * 16 + n;
          const float z = acc[r] + s_emb[eb][m * NHID + col];
          s_h[m * SHS + col] = (_Float16)sigm(z);
        }
      }
    }
    __syncthreads();

    // (C) out = h x Wout (wave -> tile); act (waves 0,1); v (wave 2)
    {
      const int n  = lane & 15;
      const int mh = (lane >> 4) << 3;

      v8f acc = {};
#pragma unroll
      for (int kt = 0; kt < 8; ++kt) {
        const v16h a = loadA(s_h, lane, kt * 32, SHS);
        const v16h b = loadBfrag(woutf + (size_t)((kt * 8 + wv) * 32) * 16, lane);
        acc = __builtin_amdgcn_wmma_f32_16x16x32_f16(false, a, false, b, (short)0, acc, false, false);
      }
#pragma unroll
      for (int r = 0; r < 8; ++r) {
        const int m = r + mh;
        out[((size_t)(b0 + m) * TLEN + t) * NCHAR + wv * 16 + n] = acc[r];
      }

      if (wv < 2) {            // action logits, pairwise softmax via shfl_xor(1)
        v8f acc2 = {};
#pragma unroll
        for (int kt = 0; kt < 8; ++kt) {
          const v16h a = loadA(s_h, lane, kt * 32, SHS);
          const v16h b = loadBfrag(&s_Wact[(kt * 2 + wv) * 32 * 16], lane);
          acc2 = __builtin_amdgcn_wmma_f32_16x16x32_f16(false, a, false, b, (short)0, acc2, false, false);
        }
#pragma unroll
        for (int r = 0; r < 8; ++r) {
          const float self  = acc2[r];
          const float other = __shfl_xor(self, 1, 32);
          const float p = __builtin_amdgcn_rcpf(1.f + __expf(other - self));
          const int m = r + mh, col = wv * 16 + n;
          if (col < 2 * NSTACK) s_act[m * 32 + col] = p;
        }
      } else if (wv == 2) {    // push values
        v8f acc2 = {};
#pragma unroll
        for (int kt = 0; kt < 8; ++kt) {
          const v16h a = loadA(s_h, lane, kt * 32, SHS);
          const v16h b = loadBfrag(&s_Wv[kt * 32 * 16], lane);
          acc2 = __builtin_amdgcn_wmma_f32_16x16x32_f16(false, a, false, b, (short)0, acc2, false, false);
        }
#pragma unroll
        for (int r = 0; r < 8; ++r) {
          const int m = r + mh;
          if (n < NSTACK) s_v[m * 16 + n] = sigm(acc2[r]);
        }
      }
    }
    __syncthreads();

    // (D) waves 0-4: soft stack shift, 4-wide f16 quads (ds_load_b64 / v_pk_fma_f16),
    //               and write the new top-2 cells straight into s_A for the next step.
    //     wave 5:   TDM gather of next step's embedding rows straight into LDS.
    if (tid < BW * NSTACK) {
      const int m = tid / NSTACK, k = tid % NSTACK;
      const _Float16 a0h = (_Float16)s_act[m * 32 + 2 * k];
      const _Float16 a1h = (_Float16)s_act[m * 32 + 2 * k + 1];
      const v4h a0v = {a0h, a0h, a0h, a0h};
      const v4h a1v = {a1h, a1h, a1h, a1h};
      v4h* row = (v4h*)&s_stacks[(m * NSTACK + k) * SSIZE];  // 50 quads, 8B aligned

      _Float16 tail = (_Float16)s_v[m * 16 + k];   // old[-1] := pushed value v
      v4h cur = row[0];
      // j = 0 peeled: also publishes new top-2 to s_A
      {
        const v4h nxt = row[1];
        const v4h s1 = {tail, cur[0], cur[1], cur[2]};        // old[s-1]
        const v4h s2 = {cur[1], cur[2], cur[3], nxt[0]};      // old[s+1]
        const v4h r  = a0v * s1 + a1v * s2;
        row[0] = r;
        s_A[m * 32 + 2 * k]     = r[0];
        s_A[m * 32 + 2 * k + 1] = r[1];
        tail = cur[3]; cur = nxt;
      }
#pragma unroll 6
      for (int j = 1; j < 49; ++j) {
        const v4h nxt = row[j + 1];
        const v4h s1 = {tail, cur[0], cur[1], cur[2]};
        const v4h s2 = {cur[1], cur[2], cur[3], nxt[0]};
        row[j] = a0v * s1 + a1v * s2;
        tail = cur[3]; cur = nxt;
      }
      { // last quad: pop tail is zero
        const v4h s1 = {tail, cur[0], cur[1], cur[2]};
        const v4h s2 = {cur[1], cur[2], cur[3], (_Float16)0.f};
        row[49] = a0v * s1 + a1v * s2;
      }
    }
#ifdef TDM_OK
    else if (wv == 5) {
      const int tt = t + 1;
      if (tt < TLEN) {
        // 16 data-dependent row indices (uniform scalar loads of x), packed 16-bit
        unsigned idx[8];
#pragma unroll
        for (int j = 0; j < 8; ++j) {
          const unsigned lo = (unsigned)x[(size_t)(b0 + 2 * j    ) * TLEN + tt] & 0xffffu;
          const unsigned hi = (unsigned)x[(size_t)(b0 + 2 * j + 1) * TLEN + tt] & 0xffffu;
          idx[j] = lo | (hi << 16);
        }
        // D# group 0: count=1, gather_mode=1 (16-bit idx), lds_addr, global_addr, type=2
        const unsigned long long ga = (unsigned long long)(size_t)embT;
        v4u g0;
        g0[0] = 0x80000001u;
        g0[1] = (unsigned)(size_t)(&s_emb[tt & 1][0]);        // LDS byte address
        g0[2] = (unsigned)ga;
        g0[3] = ((unsigned)(ga >> 32) & 0x01FFFFFFu) | 0x80000000u;
        // D# group 1: data_size=4B, tensor NCHARxNHID, tile = 16 rows x NHID
        v8i g1;
        g1[0] = (int)(2u << 16);               // data_size = 4 bytes
        g1[1] = (int)((unsigned)NHID  << 16);  // tensor_dim0 (low16 @ bits 63:48)
        g1[2] = (int)((unsigned)NCHAR << 16);  // tensor_dim1 (low16 @ bits 95:80)
        g1[3] = (int)((unsigned)NHID  << 16);  // tile_dim0   (bits 127:112)
        g1[4] = 16;                            // tile_dim1 = #valid indices
        g1[5] = NHID;                          // tensor_dim0_stride
        g1[6] = 0; g1[7] = 0;
        // D# groups 2,3: row indices
        v4i g2, g3;
        g2[0] = (int)idx[0]; g2[1] = (int)idx[1]; g2[2] = (int)idx[2]; g2[3] = (int)idx[3];
        g3[0] = (int)idx[4]; g3[1] = (int)idx[5]; g3[2] = (int)idx[6]; g3[3] = (int)idx[7];
#if __clang_major__ >= 23
        v8i g1x = {0, 0, 0, 0, 0, 0, 0, 0};
        __builtin_amdgcn_tensor_load_to_lds(g0, g1, g2, g3, g1x, 0);
#else
        __builtin_amdgcn_tensor_load_to_lds(g0, g1, g2, g3, 0);
#endif
#if __has_builtin(__builtin_amdgcn_s_wait_tensorcnt)
        __builtin_amdgcn_s_wait_tensorcnt(0);
#else
        asm volatile("s_wait_tensorcnt 0x0" ::: "memory");
#endif
      }
    } else if (wv == 6) {
      if (t + 2 < TLEN)   // keep next x column warm (global_prefetch_b8)
        __builtin_prefetch(&x[(size_t)(b0 + (lane & 15)) * TLEN + t + 2], 0, 1);
    }
#else
    else {  // fallback: manual gather by waves 5-7
      const int tt = t + 1;
      if (tt < TLEN) {
        float* dst = s_emb[tt & 1];
        for (int e = tid - BW * NSTACK; e < BW * NHID; e += 256 - BW * NSTACK) {
          const int m = e >> 8, h = e & 255;
          const int c = x[(b0 + m) * TLEN + tt];
          dst[m * NHID + h] = in2hid[h * NCHAR + c];
        }
      }
    }
#endif
    __syncthreads();
  }
}

extern "C" void kernel_launch(void* const* d_in, const int* in_sizes, int n_in,
                              void* d_out, int out_size, void* d_ws, size_t ws_size,
                              hipStream_t stream) {
  (void)in_sizes; (void)n_in; (void)out_size; (void)ws_size;
  const int*   x         = (const int*)  d_in[0];
  const float* in2hid    = (const float*)d_in[1];
  const float* hid2act   = (const float*)d_in[2];
  const float* hid2stack = (const float*)d_in[3];
  const float* stack2hid = (const float*)d_in[4];
  const float* hid2out   = (const float*)d_in[5];

  _Float16* wsf  = (_Float16*)d_ws;                       // 64 KB packed hid2out frags
  float*    embT = (float*)((char*)d_ws + 65536);         // 128 KB transposed in2hid

  stackrnn_pack_wout<<<128, 256, 0, stream>>>(hid2out, wsf);
  stackrnn_pack_embT<<<128, 256, 0, stream>>>(in2hid, embT);
  stackrnn_scan<<<NWG, 256, 0, stream>>>(x, in2hid, hid2act, hid2stack, stack2hid,
                                         wsf, embT, (float*)d_out);
}
